// BinaryTreeComposer_12180527251605
// MI455X (gfx1250) — matrix operand order repaired and checked by reading, not compiled, not run on hardware
//
#include <hip/hip_runtime.h>

#define D_DIM  2048
#define BATCH  4096
#define K_DIM  4096   // 2*D
#define N_DIM  8192   // 4*D
#define KTILES (K_DIM / 32)

typedef __attribute__((ext_vector_type(16))) __bf16 bf16x16;
typedef __attribute__((ext_vector_type(8)))  float  f32x8;

union FragU { bf16x16 v; uint4 q[2]; };

__device__ __forceinline__ float fast_sigmoid(float x) {
  return __builtin_amdgcn_rcpf(1.0f + __expf(-x));
}
__device__ __forceinline__ float fast_tanh(float x) {
#if __has_builtin(__builtin_amdgcn_tanhf)
  return __builtin_amdgcn_tanhf(x);          // v_tanh_f32 (CDNA5 trans op)
#else
  return 2.0f * __builtin_amdgcn_rcpf(1.0f + __expf(-2.0f * x)) - 1.0f;
#endif
}

// Async DMA: 16 bytes global -> LDS, tracked with ASYNCcnt (CDNA5).
__device__ __forceinline__ void async_ld_b128(unsigned lds, const void* gaddr) {
  asm volatile("global_load_async_to_lds_b128 %0, %1, off"
               :: "v"(lds), "v"(gaddr) : "memory");
}
__device__ __forceinline__ void wait_async6() {
  asm volatile("s_wait_asynccnt 6" ::: "memory");
}
__device__ __forceinline__ void wait_async0() {
  asm volatile("s_wait_asynccnt 0" ::: "memory");
}
// Low 32 bits of a flat pointer to __shared__ == wave-relative LDS offset.
__device__ __forceinline__ unsigned lds_off(const void* p) {
  return (unsigned)(uintptr_t)p;
}
// Scheduling shape: 20 DS-reads first, then 16 WMMAs back-to-back.
__device__ __forceinline__ void sched_tile_shape() {
#if __has_builtin(__builtin_amdgcn_sched_group_barrier)
  __builtin_amdgcn_sched_group_barrier(0x100, 20, 0);  // DS read x20
  __builtin_amdgcn_sched_group_barrier(0x008, 16, 0);  // MFMA/WMMA x16
#endif
}

// ---------------------------------------------------------------------------
// Pack [lh | rh] (fp32, each [4096][2048]) into bf16 A[4096][4096]
// ---------------------------------------------------------------------------
__global__ __launch_bounds__(256) void conv_a_kernel(
    const float* __restrict__ lh, const float* __restrict__ rh,
    __bf16* __restrict__ A)
{
  size_t base = ((size_t)blockIdx.x * 256 + threadIdx.x) * 8;
  size_t row  = base >> 12;
  size_t col  = base & 4095;
  const float* src = (col < (size_t)D_DIM)
      ? (lh + row * D_DIM + col)
      : (rh + row * D_DIM + (col - D_DIM));
  float4 f0 = ((const float4*)src)[0];
  float4 f1 = ((const float4*)src)[1];
  union { __bf16 o[8]; uint4 q; } u;
  u.o[0] = (__bf16)f0.x; u.o[1] = (__bf16)f0.y;
  u.o[2] = (__bf16)f0.z; u.o[3] = (__bf16)f0.w;
  u.o[4] = (__bf16)f1.x; u.o[5] = (__bf16)f1.y;
  u.o[6] = (__bf16)f1.z; u.o[7] = (__bf16)f1.w;
  *(uint4*)(A + base) = u.q;
}

// ---------------------------------------------------------------------------
// Transpose+convert [[Wl];[Wr]] (fp32 [4096][8192]) -> bf16 Wt[8192][4096]
// ---------------------------------------------------------------------------
__global__ __launch_bounds__(256) void conv_w_kernel(
    const float* __restrict__ Wl, const float* __restrict__ Wr,
    __bf16* __restrict__ Wt)
{
  __shared__ __bf16 tile[32][33];
  const int tk = blockIdx.x * 32;
  const int tn = blockIdx.y * 32;
  const int tx = threadIdx.x;
  const int ty = threadIdx.y;
#pragma unroll
  for (int j = 0; j < 32; j += 8) {
    int k = tk + ty + j;
    const float* src = (k < D_DIM)
        ? (Wl + (size_t)k * N_DIM + tn + tx)
        : (Wr + (size_t)(k - D_DIM) * N_DIM + tn + tx);
    tile[ty + j][tx] = (__bf16)(*src);
  }
  __syncthreads();
#pragma unroll
  for (int j = 0; j < 32; j += 8) {
    int n = tn + ty + j;
    Wt[(size_t)n * K_DIM + tk + tx] = tile[tx][ty + j];
  }
}

// ---------------------------------------------------------------------------
// Fused GEMM (bf16 WMMA, fp32 accum) + gate epilogue.
// Block: 256 thr = 8 waves (4x2). Block tile 128 rows x 64 gate-cols.
// Wave tile: 32x32 per gate => 16 accumulators. K loop unrolled x2 (static
// double-buffer index); sched_group_barrier pins "20 DS reads then 16 WMMAs"
// so the WMMA chain runs behind a single dscnt wait.
// ---------------------------------------------------------------------------
__global__ __launch_bounds__(256) void fused_treelstm_kernel(
    const __bf16* __restrict__ A,    // [4096][4096]
    const __bf16* __restrict__ Wt,   // [8192][4096] (N-major)
    const float*  __restrict__ bl,
    const float*  __restrict__ br,
    const float*  __restrict__ lc,
    const float*  __restrict__ rc,
    float* __restrict__ outC,
    float* __restrict__ outH)
{
  // padded rows: 40 halfs = 80B stride -> conflict-free b128 fragment loads
  __shared__ __align__(16) __bf16 As[2][128][40];       // 20 KB
  __shared__ __align__(16) __bf16 Bs[2][4][64][40];     // 40 KB

  const int t    = threadIdx.x;
  const int lane = t & 31;
  const int w    = t >> 5;
  const int wm   = w & 3;         // 32-row group
  const int wn   = w >> 2;        // 32-col group
  const int fm   = lane & 15;
  const int hi   = lane >> 4;

  const int mBase = blockIdx.x * 128;
  const int nBase = blockIdx.y * 64;

  f32x8 acc[2][2][4] = {};        // [msub][nsub][gate]

  // ---- per-thread async-copy coordinates (6 x b128 per thread per tile) ----
  const int aR0 = t >> 2, aR1 = 64 + (t >> 2), aC = (t & 3) * 8;
  const __bf16* gA0 = A + (size_t)(mBase + aR0) * K_DIM + aC;
  const __bf16* gA1 = A + (size_t)(mBase + aR1) * K_DIM + aC;
  const __bf16* gB[4];
  unsigned lB0[4], lB1[4];
#pragma unroll
  for (int j = 0; j < 4; ++j) {
    int s = t + 256 * j;
    int g = s >> 8, n = (s >> 2) & 63, c = (s & 3) * 8;
    gB[j] = Wt + (size_t)(g * D_DIM + nBase + n) * K_DIM + c;
    lB0[j] = lds_off(&Bs[0][g][n][c]);
    lB1[j] = lds_off(&Bs[1][g][n][c]);
  }
  const unsigned lA0[2] = { lds_off(&As[0][aR0][aC]), lds_off(&As[0][aR1][aC]) };
  const unsigned lA1[2] = { lds_off(&As[1][aR0][aC]), lds_off(&As[1][aR1][aC]) };

#define ISSUE_TILE(LA, LBv, KOFS)                                            \
  {                                                                          \
    async_ld_b128((LA)[0], gA0 + (KOFS));                                    \
    async_ld_b128((LA)[1], gA1 + (KOFS));                                    \
    async_ld_b128((LBv)[0], gB[0] + (KOFS));                                 \
    async_ld_b128((LBv)[1], gB[1] + (KOFS));                                 \
    async_ld_b128((LBv)[2], gB[2] + (KOFS));                                 \
    async_ld_b128((LBv)[3], gB[3] + (KOFS));                                 \
  }

#define COMPUTE_TILE(CB)                                                     \
  {                                                                          \
    FragU aF0, aF1;                                                          \
    {                                                                        \
      const __bf16* ap = &As[CB][wm * 32 + fm][hi * 8];                      \
      aF0.q[0] = ((const uint4*)ap)[0];                                      \
      aF0.q[1] = ((const uint4*)(ap + 16))[0];                               \
    }                                                                        \
    {                                                                        \
      const __bf16* ap = &As[CB][wm * 32 + 16 + fm][hi * 8];                 \
      aF1.q[0] = ((const uint4*)ap)[0];                                      \
      aF1.q[1] = ((const uint4*)(ap + 16))[0];                               \
    }                                                                        \
    FragU bF[4][2];                                                          \
    _Pragma("unroll") for (int g = 0; g < 4; ++g) {                          \
      _Pragma("unroll") for (int ns = 0; ns < 2; ++ns) {                     \
        const __bf16* bp = &Bs[CB][g][wn * 32 + ns * 16 + fm][hi * 8];       \
        bF[g][ns].q[0] = ((const uint4*)bp)[0];                              \
        bF[g][ns].q[1] = ((const uint4*)(bp + 16))[0];                       \
      }                                                                      \
    }                                                                        \
    _Pragma("unroll") for (int g = 0; g < 4; ++g) {                          \
      _Pragma("unroll") for (int ns = 0; ns < 2; ++ns) {                     \
        acc[0][ns][g] = __builtin_amdgcn_wmma_f32_16x16x32_bf16(             \
            false, aF0.v, false, bF[g][ns].v, (short)0, acc[0][ns][g],       \
            false, false);                                                   \
        acc[1][ns][g] = __builtin_amdgcn_wmma_f32_16x16x32_bf16(             \
            false, aF1.v, false, bF[g][ns].v, (short)0, acc[1][ns][g],       \
            false, true);  /* same B as previous wmma -> reuse hint */       \
      }                                                                      \
    }                                                                        \
    sched_tile_shape();                                                      \
  }

  // ---- prologue: tile 0 -> buf0 ----
  ISSUE_TILE(lA0, lB0, 0)

#pragma unroll 1
  for (int it = 0; it < KTILES; it += 2) {
    // even half: prefetch tile it+1 -> buf1, compute buf0
    ISSUE_TILE(lA1, lB1, (it + 1) * 32)     // it+1 <= KTILES-1 always
    wait_async6();                          // my tile `it` landed
    __syncthreads();                        // everyone's tile `it` landed
    COMPUTE_TILE(0)
    __syncthreads();                        // buf0 free for overwrite

    // odd half: prefetch tile it+2 -> buf0, compute buf1
    if (it + 2 < KTILES) {
      ISSUE_TILE(lA0, lB0, (it + 2) * 32)
      wait_async6();
    } else {
      wait_async0();
    }
    __syncthreads();
    COMPUTE_TILE(1)
    __syncthreads();
  }
#undef ISSUE_TILE
#undef COMPUTE_TILE

  // --------------------------- epilogue ---------------------------
#pragma unroll
  for (int ms = 0; ms < 2; ++ms) {
#pragma unroll
    for (int ns = 0; ns < 2; ++ns) {
      const int n  = nBase + wn * 32 + ns * 16 + fm;
      const int m0 = mBase + wm * 32 + ms * 16 + hi * 8;
      const float bI = bl[0 * D_DIM + n] + br[0 * D_DIM + n];
      const float bL = bl[1 * D_DIM + n] + br[1 * D_DIM + n];
      const float bR = bl[2 * D_DIM + n] + br[2 * D_DIM + n];
      const float bU = bl[3 * D_DIM + n] + br[3 * D_DIM + n];
#pragma unroll
      for (int r = 0; r < 8; ++r) {
        size_t idx = (size_t)(m0 + r) * D_DIM + n;
        float gi  = fast_sigmoid(acc[ms][ns][0][r] + bI);
        float glf = fast_sigmoid(acc[ms][ns][1][r] + bL);
        float grf = fast_sigmoid(acc[ms][ns][2][r] + bR);
        float gu  = fast_tanh   (acc[ms][ns][3][r] + bU);
        float c = gi * gu + glf * lc[idx] + grf * rc[idx];
        outC[idx] = c;
        outH[idx] = fast_tanh(c);
      }
    }
  }
}

// ---------------------------------------------------------------------------
extern "C" void kernel_launch(void* const* d_in, const int* in_sizes, int n_in,
                              void* d_out, int out_size, void* d_ws, size_t ws_size,
                              hipStream_t stream) {
  const float* lc = (const float*)d_in[0];
  const float* lh = (const float*)d_in[1];
  const float* rc = (const float*)d_in[2];
  const float* rh = (const float*)d_in[3];
  const float* Wl = (const float*)d_in[4];
  const float* bl = (const float*)d_in[5];
  const float* Wr = (const float*)d_in[6];
  const float* br = (const float*)d_in[7];

  float* outC = (float*)d_out;
  float* outH = outC + (size_t)BATCH * D_DIM;

  __bf16* Abf = (__bf16*)d_ws;                                      // 32 MiB
  __bf16* Wt  = (__bf16*)((char*)d_ws + (size_t)BATCH * K_DIM * 2); // 64 MiB

  conv_a_kernel<<<(BATCH * K_DIM / 8) / 256, 256, 0, stream>>>(lh, rh, Abf);
  conv_w_kernel<<<dim3(K_DIM / 32, N_DIM / 32), dim3(32, 8), 0, stream>>>(Wl, Wr, Wt);
  fused_treelstm_kernel<<<dim3(BATCH / 128, D_DIM / 64), 256, 0, stream>>>(
      Abf, Wt, bl, br, lc, rc, outC, outH);
}
